// MultiHeadAttention_77249281786483
// MI455X (gfx1250) — matrix operand order repaired and checked by reading, not compiled, not run on hardware
//
#include <hip/hip_runtime.h>

// ---------------- problem constants ----------------
constexpr int B_   = 2;
constexpr int S_   = 2048;
constexpr int D_   = 2048;
constexpr int H_   = 32;
constexpr int KVH_ = 8;
constexpr int HD_  = 64;     // head dim
constexpr int HALF = HD_ / 2;

typedef __attribute__((ext_vector_type(16))) __bf16 v16bf;
typedef __attribute__((ext_vector_type(8)))  float  v8f;

struct U4 { unsigned int x, y, z, w; };           // POD 16-byte chunk
union BFrag { v16bf v; U4 u[2]; };                // 16 bf16 WMMA operand fragment

// ---------------- CDNA5 async LDS DMA helpers ----------------
// GLOBAL_LOAD_ASYNC_TO_LDS_B128: per-lane 16B global -> LDS, tracked by ASYNCcnt.
__device__ __forceinline__ void async_g2l_b128(unsigned lds_byte_off, const void* gaddr) {
    asm volatile("global_load_async_to_lds_b128 %0, %1, off"
                 :: "v"(lds_byte_off), "v"(gaddr)
                 : "memory");
}
__device__ __forceinline__ void wait_async0() {
    asm volatile("s_wait_asynccnt 0x0" ::: "memory");
}

// ---------------- elementwise conversion kernels ----------------
__global__ void cvt_f32_bf16(const float* __restrict__ src, __bf16* __restrict__ dst, int n) {
    int i = blockIdx.x * blockDim.x + threadIdx.x;
    if (i < n) dst[i] = (__bf16)src[i];
}

// dst (N x K, row-major bf16) = transpose of src (K x N, row-major f32)
__global__ void cvtT_f32_bf16(const float* __restrict__ src, __bf16* __restrict__ dst, int K, int N) {
    int i = blockIdx.x * blockDim.x + threadIdx.x;
    if (i >= N * K) return;
    int n = i / K, k = i % K;
    dst[(size_t)n * K + k] = (__bf16)src[(size_t)k * N + n];
}

// ---------------- RoPE / layout kernels ----------------
__global__ void rope_q_kernel(const float* __restrict__ Qf, __bf16* __restrict__ Qb,
                              const float* __restrict__ cosT, const float* __restrict__ sinT) {
    int i = blockIdx.x * blockDim.x + threadIdx.x;
    if (i >= B_ * S_ * H_ * HALF) return;
    int p = i % HALF; int t = i / HALF;
    int h = t % H_;   t /= H_;
    int s = t % S_;   int b = t / S_;
    size_t base = ((size_t)(b * S_ + s)) * D_ + h * HD_ + 2 * p;
    float t1 = Qf[base], t2 = Qf[base + 1];
    float c = cosT[s * HALF + p], sn = sinT[s * HALF + p];
    Qb[base]     = (__bf16)(t1 * c - t2 * sn);
    Qb[base + 1] = (__bf16)(t1 * sn + t2 * c);
}

__global__ void rope_k_kernel(const float* __restrict__ Kf, __bf16* __restrict__ Kb,
                              const float* __restrict__ cosT, const float* __restrict__ sinT) {
    int i = blockIdx.x * blockDim.x + threadIdx.x;
    if (i >= B_ * S_ * KVH_ * HALF) return;
    int p = i % HALF; int t = i / HALF;
    int kh = t % KVH_; t /= KVH_;
    int s = t % S_;    int b = t / S_;
    size_t sbase = ((size_t)(b * S_ + s)) * (KVH_ * HD_) + kh * HD_ + 2 * p;
    float t1 = Kf[sbase], t2 = Kf[sbase + 1];
    float c = cosT[s * HALF + p], sn = sinT[s * HALF + p];
    size_t dbase = (((size_t)(b * KVH_ + kh)) * S_ + s) * HD_ + 2 * p;
    Kb[dbase]     = (__bf16)(t1 * c - t2 * sn);
    Kb[dbase + 1] = (__bf16)(t1 * sn + t2 * c);
}

__global__ void v_transpose_kernel(const float* __restrict__ Vf, __bf16* __restrict__ Vt) {
    int i = blockIdx.x * blockDim.x + threadIdx.x;
    if (i >= B_ * S_ * KVH_ * HD_) return;
    int d = i % HD_;  int t = i / HD_;
    int kh = t % KVH_; t /= KVH_;
    int s = t % S_;    int b = t / S_;
    Vt[(((size_t)(b * KVH_ + kh)) * HD_ + d) * S_ + s] =
        (__bf16)Vf[((size_t)(b * S_ + s)) * (KVH_ * HD_) + kh * HD_ + d];
}

// ---------------- WMMA GEMM with async-DMA double buffering ----------------
// C(f32, MxN) = A(bf16, MxK) * Bt(bf16, NxK)^T [+bias]
constexpr int TM = 128, TN = 128, TK = 32, LST = 48; // LDS row stride (16B aligned rows)

__global__ __launch_bounds__(256)
void gemm_bf16_wmma(const __bf16* __restrict__ A, const __bf16* __restrict__ Bt,
                    const float* __restrict__ bias, float* __restrict__ C,
                    int M, int N, int K) {
    __shared__ __bf16 As[2][TM][LST];
    __shared__ __bf16 Bs[2][TN][LST];
    const int tid = threadIdx.x;
    const int wave = tid >> 5, lane = tid & 31;
    const int lhalf = lane >> 4, l16 = lane & 15;
    const int wm = wave & 1, wn = wave >> 1;     // 2 waves along M, 4 along N
    const int bm = blockIdx.y * TM, bn = blockIdx.x * TN;

    // per-thread staging assignment: 32B of one row of each tile
    const int srow = tid >> 1, shalf = (tid & 1) << 4;   // element offset 0 / 16
    const __bf16* ga = A  + (size_t)(bm + srow) * K + shalf;
    const __bf16* gb = Bt + (size_t)(bn + srow) * K + shalf;
    const unsigned la0 = (unsigned)(uintptr_t)&As[0][srow][shalf];
    const unsigned lb0 = (unsigned)(uintptr_t)&Bs[0][srow][shalf];
    constexpr unsigned ABUF = TM * LST * 2;              // bytes per buffer
    constexpr unsigned BBUF = TN * LST * 2;

    auto stage = [&](int buf, int k0) {
        const __bf16* pa = ga + k0;
        const __bf16* pb = gb + k0;
        unsigned la = la0 + (unsigned)buf * ABUF;
        unsigned lb = lb0 + (unsigned)buf * BBUF;
        async_g2l_b128(la,      pa);
        async_g2l_b128(la + 16, pa + 8);
        async_g2l_b128(lb,      pb);
        async_g2l_b128(lb + 16, pb + 8);
    };

    v8f c[4][2];
#pragma unroll
    for (int mi = 0; mi < 4; ++mi)
#pragma unroll
        for (int ni = 0; ni < 2; ++ni) {
            v8f z = {0.f, 0.f, 0.f, 0.f, 0.f, 0.f, 0.f, 0.f};
            c[mi][ni] = z;
        }

    // prologue: DMA first tile
    stage(0, 0);
    wait_async0();
    __syncthreads();

    int cur = 0;
    for (int k0 = 0; k0 < K; k0 += TK) {
        // kick DMA for next tile into the other buffer (overlaps with WMMA below)
        if (k0 + TK < K) {
            stage(cur ^ 1, k0 + TK);
            __builtin_prefetch(ga + k0 + 2 * TK, 0, 3);   // warm L2 two tiles ahead
            __builtin_prefetch(gb + k0 + 2 * TK, 0, 3);
        }

        BFrag a[4], b[2];
#pragma unroll
        for (int mi = 0; mi < 4; ++mi) {
            int m = wm * 64 + mi * 16 + l16;
            a[mi].u[0] = *(const U4*)&As[cur][m][lhalf * 8];
            a[mi].u[1] = *(const U4*)&As[cur][m][16 + lhalf * 8];
        }
#pragma unroll
        for (int ni = 0; ni < 2; ++ni) {
            int n = wn * 32 + ni * 16 + l16;
            b[ni].u[0] = *(const U4*)&Bs[cur][n][lhalf * 8];
            b[ni].u[1] = *(const U4*)&Bs[cur][n][16 + lhalf * 8];
        }
#pragma unroll
        for (int mi = 0; mi < 4; ++mi)
#pragma unroll
            for (int ni = 0; ni < 2; ++ni)
                c[mi][ni] = __builtin_amdgcn_wmma_f32_16x16x32_bf16(
                    false, a[mi].v, false, b[ni].v, (short)0, c[mi][ni], false, false);

        // own DMA done + all waves done reading/writing before swap
        wait_async0();
        __syncthreads();
        cur ^= 1;
    }

#pragma unroll
    for (int mi = 0; mi < 4; ++mi)
#pragma unroll
        for (int ni = 0; ni < 2; ++ni) {
            int col = bn + wn * 32 + ni * 16 + l16;
            float bv = bias ? bias[col] : 0.0f;
#pragma unroll
            for (int r = 0; r < 8; ++r) {
                int row = bm + wm * 64 + mi * 16 + lhalf * 8 + r;
                C[(size_t)row * N + col] = c[mi][ni][r] + bv;
            }
        }
}

// ---------------- flash attention (causal, GQA, online softmax) ----------------
// grid: (S/64, H, B); block: 128 threads = 4 waves; each wave owns 16 q rows.
__global__ __launch_bounds__(128)
void flash_attn_kernel(const __bf16* __restrict__ Qb, const __bf16* __restrict__ Kb,
                       const __bf16* __restrict__ Vt, __bf16* __restrict__ ctx) {
    __shared__ __bf16 p_lds[4][16][32];   // per-wave P reshape slab
    const int tid = threadIdx.x;
    const int wave = tid >> 5, lane = tid & 31;
    const int lhalf = lane >> 4, l16 = lane & 15;
    const int b = blockIdx.z, h = blockIdx.y, qb = blockIdx.x;
    const int kvh = h >> 2;                       // N_REP = 4
    const int qrow = qb * 64 + wave * 16;

    // Q A-fragments (16 rows x 64 dims -> 2 k-chunks of 32)
    BFrag aQ[2];
    {
        const __bf16* qp = Qb + ((size_t)(b * S_ + qrow + l16)) * D_ + h * HD_;
#pragma unroll
        for (int kc = 0; kc < 2; ++kc) {
            aQ[kc].u[0] = *(const U4*)(qp + kc * 32 + lhalf * 8);
            aQ[kc].u[1] = *(const U4*)(qp + kc * 32 + 16 + lhalf * 8);
        }
    }

    v8f acc[4];
#pragma unroll
    for (int d = 0; d < 4; ++d) {
        v8f z = {0.f, 0.f, 0.f, 0.f, 0.f, 0.f, 0.f, 0.f};
        acc[d] = z;
    }
    float mrow[8], lrow[8];
#pragma unroll
    for (int r = 0; r < 8; ++r) { mrow[r] = -1e30f; lrow[r] = 0.f; }

    const __bf16* kball = Kb + ((size_t)(b * KVH_ + kvh)) * S_ * HD_;
    const __bf16* vball = Vt + ((size_t)(b * KVH_ + kvh)) * HD_ * S_;

    const int nkb = 2 * qb + 2;                   // causal: keys 0 .. qb*64+63
    for (int kb = 0; kb < nkb; ++kb) {
        const int kbase = kb * 32;

        // K B-fragments: 2 key-chunks x 2 kdim-chunks
        BFrag bK[2][2];
#pragma unroll
        for (int ch = 0; ch < 2; ++ch) {
            const __bf16* kp = kball + ((size_t)(kbase + ch * 16 + l16)) * HD_;
#pragma unroll
            for (int kc = 0; kc < 2; ++kc) {
                bK[ch][kc].u[0] = *(const U4*)(kp + kc * 32 + lhalf * 8);
                bK[ch][kc].u[1] = *(const U4*)(kp + kc * 32 + 16 + lhalf * 8);
            }
        }

        // scores: 16q x 32k
        v8f sc[2];
#pragma unroll
        for (int ch = 0; ch < 2; ++ch) {
            v8f z = {0.f, 0.f, 0.f, 0.f, 0.f, 0.f, 0.f, 0.f};
            sc[ch] = z;
#pragma unroll
            for (int kc = 0; kc < 2; ++kc)
                sc[ch] = __builtin_amdgcn_wmma_f32_16x16x32_bf16(
                    false, aQ[kc].v, false, bK[ch][kc].v, (short)0, sc[ch], false, false);
        }

        // online softmax (rows live across the 16 lanes of each half-wave)
        float pv0[8], pv1[8];
#pragma unroll
        for (int r = 0; r < 8; ++r) {
            int row = qrow + lhalf * 8 + r;
            int c0 = kbase + l16, c1 = kbase + 16 + l16;
            float v0 = sc[0][r] * 0.125f;         // 1/sqrt(64)
            float v1 = sc[1][r] * 0.125f;
            if (c0 > row) v0 = -1e30f;
            if (c1 > row) v1 = -1e30f;
            float vmax = fmaxf(v0, v1);
#pragma unroll
            for (int o = 8; o >= 1; o >>= 1) vmax = fmaxf(vmax, __shfl_xor(vmax, o, 32));
            float mnew = fmaxf(mrow[r], vmax);
            float scale = __expf(mrow[r] - mnew);
            float p0 = __expf(v0 - mnew), p1 = __expf(v1 - mnew);
            float rs = p0 + p1;
#pragma unroll
            for (int o = 8; o >= 1; o >>= 1) rs += __shfl_xor(rs, o, 32);
            lrow[r] = lrow[r] * scale + rs;
            mrow[r] = mnew;
#pragma unroll
            for (int d = 0; d < 4; ++d) acc[d][r] = acc[d][r] * scale;
            pv0[r] = p0; pv1[r] = p1;
        }

        // reshape P: C-fragment layout -> A-fragment layout via per-wave LDS
        __syncthreads();                           // uniform across waves (nkb same per block)
#pragma unroll
        for (int r = 0; r < 8; ++r) {
            p_lds[wave][lhalf * 8 + r][l16]      = (__bf16)pv0[r];
            p_lds[wave][lhalf * 8 + r][16 + l16] = (__bf16)pv1[r];
        }
        __syncthreads();
        BFrag aP;
        aP.u[0] = *(const U4*)&p_lds[wave][l16][lhalf * 8];
        aP.u[1] = *(const U4*)&p_lds[wave][l16][16 + lhalf * 8];

        // ctx += P * V   (V transposed: dim-major, keys contiguous)
#pragma unroll
        for (int d = 0; d < 4; ++d) {
            BFrag bV;
            const __bf16* p = vball + ((size_t)(d * 16 + l16)) * S_ + kbase + lhalf * 8;
            bV.u[0] = *(const U4*)p; bV.u[1] = *(const U4*)(p + 16);
            acc[d] = __builtin_amdgcn_wmma_f32_16x16x32_bf16(
                false, aP.v, false, bV.v, (short)0, acc[d], false, false);
        }
    }

    // normalize + store bf16 ctx in [b,s,D] layout for output projection
#pragma unroll
    for (int d = 0; d < 4; ++d)
#pragma unroll
        for (int r = 0; r < 8; ++r) {
            int row = qrow + lhalf * 8 + r;
            int col = h * HD_ + d * 16 + l16;
            ctx[((size_t)(b * S_ + row)) * D_ + col] = (__bf16)(acc[d][r] / lrow[r]);
        }
}

// ---------------- host: workspace layout + launch sequence ----------------
extern "C" void kernel_launch(void* const* d_in, const int* in_sizes, int n_in,
                              void* d_out, int out_size, void* d_ws, size_t ws_size,
                              hipStream_t stream) {
    const float* x    = (const float*)d_in[0];
    const float* Wq   = (const float*)d_in[1];
    const float* Wk   = (const float*)d_in[2];
    const float* Wv   = (const float*)d_in[3];
    const float* Wo   = (const float*)d_in[4];
    const float* bo   = (const float*)d_in[5];
    const float* cosT = (const float*)d_in[6];
    const float* sinT = (const float*)d_in[7];
    float* out = (float*)d_out;

    char* ws = (char*)d_ws;
    size_t off = 0;
    auto alloc = [&](size_t bytes) { void* p = ws + off; off += (bytes + 255) & ~size_t(255); return p; };
    __bf16* xb  = (__bf16*)alloc((size_t)B_ * S_ * D_ * 2);
    __bf16* wqT = (__bf16*)alloc((size_t)D_ * D_ * 2);
    __bf16* wkT = (__bf16*)alloc((size_t)D_ * (KVH_ * HD_) * 2);
    __bf16* wvT = (__bf16*)alloc((size_t)D_ * (KVH_ * HD_) * 2);
    __bf16* woT = (__bf16*)alloc((size_t)D_ * D_ * 2);
    float*  Qf  = (float*) alloc((size_t)B_ * S_ * D_ * 4);
    float*  Kf  = (float*) alloc((size_t)B_ * S_ * (KVH_ * HD_) * 4);
    float*  Vf  = (float*) alloc((size_t)B_ * S_ * (KVH_ * HD_) * 4);
    __bf16* Qbb = (__bf16*)alloc((size_t)B_ * S_ * D_ * 2);
    __bf16* Kbb = (__bf16*)alloc((size_t)B_ * KVH_ * S_ * HD_ * 2);
    __bf16* Vtb = (__bf16*)alloc((size_t)B_ * KVH_ * HD_ * S_ * 2);
    __bf16* ctx = (__bf16*)alloc((size_t)B_ * S_ * D_ * 2);

    const int M = B_ * S_;          // 4096
    const int nX = B_ * S_ * D_;    // 8388608

    // 1. convert / transpose-convert
    cvt_f32_bf16<<<(nX + 255) / 256, 256, 0, stream>>>(x, xb, nX);
    cvtT_f32_bf16<<<(D_ * D_ + 255) / 256, 256, 0, stream>>>(Wq, wqT, D_, D_);
    cvtT_f32_bf16<<<(D_ * KVH_ * HD_ + 255) / 256, 256, 0, stream>>>(Wk, wkT, D_, KVH_ * HD_);
    cvtT_f32_bf16<<<(D_ * KVH_ * HD_ + 255) / 256, 256, 0, stream>>>(Wv, wvT, D_, KVH_ * HD_);
    cvtT_f32_bf16<<<(D_ * D_ + 255) / 256, 256, 0, stream>>>(Wo, woT, D_, D_);

    // 2. QKV projections (WMMA + async DMA double buffering)
    dim3 gq(D_ / TN, M / TM);                 // (16, 32)
    dim3 gkv((KVH_ * HD_) / TN, M / TM);      // (4, 32)
    gemm_bf16_wmma<<<gq,  256, 0, stream>>>(xb, wqT, nullptr, Qf, M, D_,         D_);
    gemm_bf16_wmma<<<gkv, 256, 0, stream>>>(xb, wkT, nullptr, Kf, M, KVH_ * HD_, D_);
    gemm_bf16_wmma<<<gkv, 256, 0, stream>>>(xb, wvT, nullptr, Vf, M, KVH_ * HD_, D_);

    // 3. RoPE + attention layouts
    int nQp = B_ * S_ * H_ * HALF, nKp = B_ * S_ * KVH_ * HALF, nV = B_ * S_ * KVH_ * HD_;
    rope_q_kernel<<<(nQp + 255) / 256, 256, 0, stream>>>(Qf, Qbb, cosT, sinT);
    rope_k_kernel<<<(nKp + 255) / 256, 256, 0, stream>>>(Kf, Kbb, cosT, sinT);
    v_transpose_kernel<<<(nV + 255) / 256, 256, 0, stream>>>(Vf, Vtb);

    // 4. causal flash attention (WMMA)
    dim3 gf(S_ / 64, H_, B_);
    flash_attn_kernel<<<gf, 128, 0, stream>>>(Qbb, Kbb, Vtb, ctx);

    // 5. output projection + bias (WMMA), fp32 out
    gemm_bf16_wmma<<<gq, 256, 0, stream>>>(ctx, woT, bo, out, M, D_, D_);
}
